// GraphAttentionLayer_65541200937615
// MI455X (gfx1250) — matrix operand order repaired and checked by reading, not compiled, not run on hardware
//
#include <hip/hip_runtime.h>
#include <hip/hip_bf16.h>

typedef __attribute__((ext_vector_type(16))) _Float16 v16h;
typedef __attribute__((ext_vector_type(8)))  _Float16 v8h;
typedef __attribute__((ext_vector_type(8)))  float    v8f;

#define B_     8
#define N_     2048
#define FIN_   768
#define FOUT_  64
#define ALPHA_ 0.2f
#define NEGINF_ -9.0e15f

// ---------------------------------------------------------------------------
// W [768][64] f32  ->  WT [64][768] f16  (so B-fragments of the Wh GEMM are
// contiguous 16B loads along K)
// ---------------------------------------------------------------------------
__global__ void gat_wt_kernel(const float* __restrict__ W, _Float16* __restrict__ WT) {
    int idx = blockIdx.x * blockDim.x + threadIdx.x;
    if (idx >= FIN_ * FOUT_) return;
    int k = idx / FOUT_, f = idx % FOUT_;
    WT[(size_t)f * FIN_ + k] = (_Float16)W[idx];
}

// ---------------------------------------------------------------------------
// Wh = h @ W via v_wmma_f32_16x16x32_f16. Each wave: 16 rows x 64 cols.
// Output stored TRANSPOSED as f16: WhT[b][f][i]  (feeds e-kernel and the
// B-fragments of the attention GEMM with contiguous loads).
// ---------------------------------------------------------------------------
__global__ void __launch_bounds__(128)
gat_wh_gemm_kernel(const float* __restrict__ h, const _Float16* __restrict__ WT,
                   _Float16* __restrict__ WhT) {
    const int wave = threadIdx.x >> 5, lane = threadIdx.x & 31;
    const int m  = lane & 15;       // A-row / B-col owned by this lane
    const int hh = lane >> 4;       // lane half-group
    const int i0 = blockIdx.x * 64 + wave * 16;      // global row (over B*N)
    const float* hrow = h + (size_t)(i0 + m) * FIN_;

    v8f c[4] = {};                                    // 16x64 accumulator
    for (int kb = 0; kb < FIN_; kb += 32) {
        // Issue ALL loads for this K-step up front (A: 4x16B, B: 4x32B),
        // then convert + drain the 4 WMMAs back-to-back (no D->A/B hazard).
        const float4 a0 = *(const float4*)(hrow + kb + 8 * hh);
        const float4 a1 = *(const float4*)(hrow + kb + 8 * hh + 4);
        const float4 a2 = *(const float4*)(hrow + kb + 16 + 8 * hh);
        const float4 a3 = *(const float4*)(hrow + kb + 16 + 8 * hh + 4);
        v16h bf[4];
#pragma unroll
        for (int t = 0; t < 4; ++t)
            bf[t] = __builtin_shufflevector(*(const v8h*)(WT + (size_t)(t * 16 + m) * FIN_ + kb + 8 * hh), *(const v8h*)(WT + (size_t)(t * 16 + m) * FIN_ + kb + 16 + 8 * hh), 0,1,2,3,4,5,6,7,8,9,10,11,12,13,14,15);

        v16h af;
        af[0]=(_Float16)a0.x; af[1]=(_Float16)a0.y; af[2]=(_Float16)a0.z; af[3]=(_Float16)a0.w;
        af[4]=(_Float16)a1.x; af[5]=(_Float16)a1.y; af[6]=(_Float16)a1.z; af[7]=(_Float16)a1.w;
        af[8]=(_Float16)a2.x; af[9]=(_Float16)a2.y; af[10]=(_Float16)a2.z; af[11]=(_Float16)a2.w;
        af[12]=(_Float16)a3.x; af[13]=(_Float16)a3.y; af[14]=(_Float16)a3.z; af[15]=(_Float16)a3.w;
#pragma unroll
        for (int t = 0; t < 4; ++t)
            c[t] = __builtin_amdgcn_wmma_f32_16x16x32_f16(
                false, af, false, bf[t], (short)0, c[t], false, false);
    }
    // C layout: VGPR v holds row (v + 8*hh), col (t*16 + m).
    const int bb = i0 / N_;
    const int ilbase = (i0 % N_) + 8 * hh;
#pragma unroll
    for (int t = 0; t < 4; ++t) {
        const int f = t * 16 + m;
        v8h o;
#pragma unroll
        for (int v = 0; v < 8; ++v) o[v] = (_Float16)c[t][v];
        *(v8h*)(WhT + ((size_t)bb * FOUT_ + f) * N_ + ilbase) = o;   // 16B store
    }
}

// ---------------------------------------------------------------------------
// Rank-1 projections: e1[b,i] = Wh[b,i,:]·a[:64], e2[b,i] = Wh[b,i,:]·a[64:]
// ---------------------------------------------------------------------------
__global__ void gat_e_kernel(const _Float16* __restrict__ WhT, const float* __restrict__ a,
                             float* __restrict__ e1, float* __restrict__ e2) {
    int idx = blockIdx.x * blockDim.x + threadIdx.x;
    if (idx >= B_ * N_) return;
    int b = idx / N_, i = idx % N_;
    const _Float16* p = WhT + (size_t)b * FOUT_ * N_ + i;
    float s1 = 0.f, s2 = 0.f;
#pragma unroll
    for (int f = 0; f < FOUT_; ++f) {
        float wv = (float)p[(size_t)f * N_];     // coalesced across threads
        s1 += wv * a[f];
        s2 += wv * a[FOUT_ + f];
    }
    e1[idx] = s1;
    e2[idx] = s2;
}

// ---------------------------------------------------------------------------
// Column softmax stats (softmax axis=1!). One thread per column j.
// Online max+sum over rows i; emit col[j] = (e2_j, q_j = m_j + log d_j) so the
// fused kernel computes att = exp(s - q_j). Exactly matches the reference,
// including fully-masked columns (m=-9e15, d=2048 -> uniform 1/2048).
// ---------------------------------------------------------------------------
__global__ void gat_colstats_kernel(const float* __restrict__ adj, const float* __restrict__ e1,
                                    const float* __restrict__ e2, float2* __restrict__ col) {
    int idx = blockIdx.x * blockDim.x + threadIdx.x;
    if (idx >= B_ * N_) return;
    int b = idx / N_, j = idx % N_;
    const float  e2j = e2[idx];
    const float* ac  = adj + (size_t)b * N_ * N_ + j;   // column j, coalesced over threads
    const float* e1b = e1 + (size_t)b * N_;
    float mm = NEGINF_, d = 0.f;
#pragma unroll 4
    for (int i = 0; i < N_; ++i) {
        __builtin_prefetch(ac + (size_t)(i + 8) * N_, 0, 1);   // stream adj ahead
        float av = ac[(size_t)i * N_];
        float ss = e1b[i] + e2j;
        float lr = ss > 0.f ? ss : ALPHA_ * ss;
        float s  = av > 0.f ? lr : NEGINF_;
        float nm = fmaxf(mm, s);
        d  = d * __expf(mm - nm) + __expf(s - nm);
        mm = nm;
    }
    col[idx] = make_float2(e2j, mm + __logf(d));        // d >= 1 always
}

__device__ __forceinline__ _Float16 gat_prob(float av, float e2j, float qj, float e1row) {
    float ss = e1row + e2j;
    float lr = ss > 0.f ? ss : ALPHA_ * ss;
    float s  = av > 0.f ? lr : NEGINF_;
    return (_Float16)__expf(s - qj);
}

// ---------------------------------------------------------------------------
// out = ELU( att @ Wh ): fused second adj pass. Each wave: 16 rows x 64 cols.
// A fragment = attention probabilities computed on the fly (f16),
// B fragment = two 16B loads from WhT (at j0 and j1). 4 WMMAs per 16x32 j-tile.
// ---------------------------------------------------------------------------
__global__ void __launch_bounds__(128)
gat_attn_out_kernel(const float* __restrict__ adj, const _Float16* __restrict__ WhT,
                    const float* __restrict__ e1, const float2* __restrict__ col,
                    float* __restrict__ out) {
    const int b    = blockIdx.y;
    const int wave = threadIdx.x >> 5, lane = threadIdx.x & 31;
    const int m  = lane & 15;
    const int hh = lane >> 4;
    const int i0 = blockIdx.x * 64 + wave * 16;
    const int row = i0 + m;

    const float   e1row  = e1[(size_t)b * N_ + row];
    const float*  adjrow = adj + ((size_t)b * N_ + row) * N_;
    const float2* colb   = col + (size_t)b * N_;
    const _Float16* whtb = WhT + (size_t)b * FOUT_ * N_;

    v8f c[4] = {};
    for (int jb = 0; jb < N_; jb += 32) {
        const int j0 = jb + 8 * hh;           // K halves 0..7
        const int j1 = jb + 16 + 8 * hh;      // K halves 8..15
        __builtin_prefetch(adjrow + jb + 512, 0, 1);           // stream adj ahead

        // Issue all global loads for this tile up front: adj 4x16B,
        // col 8x16B, B-fragments 4x32B -> one long clause, then compute.
        const float4 ad0 = *(const float4*)(adjrow + j0);
        const float4 ad1 = *(const float4*)(adjrow + j0 + 4);
        const float4 ad2 = *(const float4*)(adjrow + j1);
        const float4 ad3 = *(const float4*)(adjrow + j1 + 4);
        const float4 q0 = *(const float4*)(colb + j0);
        const float4 q1 = *(const float4*)(colb + j0 + 2);
        const float4 q2 = *(const float4*)(colb + j0 + 4);
        const float4 q3 = *(const float4*)(colb + j0 + 6);
        const float4 q4 = *(const float4*)(colb + j1);
        const float4 q5 = *(const float4*)(colb + j1 + 2);
        const float4 q6 = *(const float4*)(colb + j1 + 4);
        const float4 q7 = *(const float4*)(colb + j1 + 6);
        v16h bf[4];
#pragma unroll
        for (int t = 0; t < 4; ++t)
            bf[t] = __builtin_shufflevector(*(const v8h*)(whtb + (size_t)(t * 16 + m) * N_ + j0), *(const v8h*)(whtb + (size_t)(t * 16 + m) * N_ + j1), 0,1,2,3,4,5,6,7,8,9,10,11,12,13,14,15);

        v16h af;
        af[0]  = gat_prob(ad0.x, q0.x, q0.y, e1row);
        af[1]  = gat_prob(ad0.y, q0.z, q0.w, e1row);
        af[2]  = gat_prob(ad0.z, q1.x, q1.y, e1row);
        af[3]  = gat_prob(ad0.w, q1.z, q1.w, e1row);
        af[4]  = gat_prob(ad1.x, q2.x, q2.y, e1row);
        af[5]  = gat_prob(ad1.y, q2.z, q2.w, e1row);
        af[6]  = gat_prob(ad1.z, q3.x, q3.y, e1row);
        af[7]  = gat_prob(ad1.w, q3.z, q3.w, e1row);
        af[8]  = gat_prob(ad2.x, q4.x, q4.y, e1row);
        af[9]  = gat_prob(ad2.y, q4.z, q4.w, e1row);
        af[10] = gat_prob(ad2.z, q5.x, q5.y, e1row);
        af[11] = gat_prob(ad2.w, q5.z, q5.w, e1row);
        af[12] = gat_prob(ad3.x, q6.x, q6.y, e1row);
        af[13] = gat_prob(ad3.y, q6.z, q6.w, e1row);
        af[14] = gat_prob(ad3.z, q7.x, q7.y, e1row);
        af[15] = gat_prob(ad3.w, q7.z, q7.w, e1row);

#pragma unroll
        for (int t = 0; t < 4; ++t)
            c[t] = __builtin_amdgcn_wmma_f32_16x16x32_f16(
                false, af, false, bf[t], (short)0, c[t], false, false);
    }
    // ELU + store. C VGPR v -> row i0 + 8*hh + v, col t*16 + m.
#pragma unroll
    for (int t = 0; t < 4; ++t) {
        const int f = t * 16 + m;
#pragma unroll
        for (int v = 0; v < 8; ++v) {
            float r = c[t][v];
            float val = r > 0.f ? r : (__expf(r) - 1.f);
            out[((size_t)b * N_ + (i0 + 8 * hh + v)) * FOUT_ + f] = val;
        }
    }
}

// ---------------------------------------------------------------------------
extern "C" void kernel_launch(void* const* d_in, const int* in_sizes, int n_in,
                              void* d_out, int out_size, void* d_ws, size_t ws_size,
                              hipStream_t stream) {
    const float* h   = (const float*)d_in[0];   // [8,2048,768]
    const float* adj = (const float*)d_in[1];   // [8,2048,2048]
    const float* W   = (const float*)d_in[2];   // [768,64]
    const float* a   = (const float*)d_in[3];   // [128,1]
    float* out = (float*)d_out;                 // [8,2048,64]
    (void)in_sizes; (void)n_in; (void)out_size; (void)ws_size;

    char*  ws  = (char*)d_ws;
    size_t off = 0;
    auto take = [&](size_t bytes) -> char* {
        off = (off + 255) & ~(size_t)255;
        char* p = ws + off;
        off += bytes;
        return p;
    };
    _Float16* WhT = (_Float16*)take((size_t)B_ * FOUT_ * N_ * sizeof(_Float16)); // 2 MB
    _Float16* WT  = (_Float16*)take((size_t)FOUT_ * FIN_ * sizeof(_Float16));
    float*    e1  = (float*)take((size_t)B_ * N_ * sizeof(float));
    float*    e2  = (float*)take((size_t)B_ * N_ * sizeof(float));
    float2*   col = (float2*)take((size_t)B_ * N_ * sizeof(float2));

    gat_wt_kernel<<<(FIN_ * FOUT_ + 255) / 256, 256, 0, stream>>>(W, WT);
    gat_wh_gemm_kernel<<<(B_ * N_) / 64, 128, 0, stream>>>(h, WT, WhT);
    gat_e_kernel<<<(B_ * N_) / 256, 256, 0, stream>>>(WhT, a, e1, e2);
    gat_colstats_kernel<<<(B_ * N_) / 256, 256, 0, stream>>>(adj, e1, e2, col);
    gat_attn_out_kernel<<<dim3(N_ / 64, B_), 128, 0, stream>>>(adj, WhT, e1, col, out);
}
